// FusedExpertsWrapper_34918084116585
// MI455X (gfx1250) — compile-verified, matrix operand order checked by reading, not run on hardware
//
#include <hip/hip_runtime.h>
#include <hip/hip_bf16.h>

// MoE expert FFN: E=8, H=2048, I=2048, tokens per expert = 8*512 = 4096.
// GEMM1 (4096x2048 @ 2048x4096) + SwiGLU -> bf16 activations in d_ws,
// GEMM2 (4096x2048 @ 2048x2048) + bias -> fp32 out (E, 4096, 2048).
// Compute-bound (~824 GFLOP vs ~40us of HBM at 23.3 TB/s), so the math runs
// on v_wmma_f32_16x16x32_bf16 with f32 accumulation. Double-buffered LDS
// tiles (one barrier per K-step); kernel2's A tile (already bf16) streams in
// via global_load_async_to_lds_b128 on the ASYNCcnt path.

typedef __attribute__((ext_vector_type(16))) __bf16 v16bf;
typedef __attribute__((ext_vector_type(4)))  __bf16 v4bf;
typedef __attribute__((ext_vector_type(8)))  float  v8f;

#define E_    8
#define H_    2048
#define I_    2048
#define F2_   4096      // 2*I (gate/up interleaved)
#define MTOT  4096      // tokens per expert (8 chunks * 512)
#define BM    128
#define BN    128
#define BK    32
#define LDT   40        // LDS tile row stride (bf16 elems): 80B, 16B-aligned rows
#define LDC   132       // LDS C row stride (floats)
#define TILEB (BM * LDT * 2)   // 10240 B per tile stage

// Load one 16x32 bf16 WMMA operand fragment from an LDS tile stored as
// [row][k] with stride LDT. Lane L: (L&15) = row/col, kb = 8*(L>>4).
// Two ds_load_b128: K {kb..kb+7} and {kb+16..kb+23} per the ISA layout.
__device__ __forceinline__ v16bf frag_ld(const __bf16* p) {
    union { uint4 u[2]; v16bf v; } c;
    c.u[0] = *(const uint4*)(p);
    c.u[1] = *(const uint4*)(p + 16);
    return c.v;
}

// ---------------- GEMM1 + SwiGLU ----------------

__device__ __forceinline__ void k1_load_tiles(const float* __restrict__ Abase,
                                              const float* __restrict__ Bbase,
                                              int kb, __bf16* sa, __bf16* sb,
                                              int tid) {
    // A tile: 128x32 fp32 -> bf16 into sa[row][k]
    #pragma unroll
    for (int i = 0; i < 4; ++i) {
        int seg = i * 256 + tid;
        int row = seg >> 3, c4 = (seg & 7) << 2;
        const float4 f = *(const float4*)(Abase + (size_t)row * H_ + kb + c4);
        v4bf h; h.x = (__bf16)f.x; h.y = (__bf16)f.y;
                h.z = (__bf16)f.z; h.w = (__bf16)f.w;
        *(v4bf*)(sa + row * LDT + c4) = h;
    }
    // B tile: 32x128 fp32, transpose-convert into sb[n][k]
    #pragma unroll
    for (int i = 0; i < 4; ++i) {
        int seg = i * 256 + tid;
        int kr = seg >> 5, fi = (seg & 31) << 2;
        const float4 f = *(const float4*)(Bbase + (size_t)(kb + kr) * F2_ + fi);
        sb[(fi + 0) * LDT + kr] = (__bf16)f.x;
        sb[(fi + 1) * LDT + kr] = (__bf16)f.y;
        sb[(fi + 2) * LDT + kr] = (__bf16)f.z;
        sb[(fi + 3) * LDT + kr] = (__bf16)f.w;
    }
}

__global__ __launch_bounds__(256) void
moe_gemm1_swiglu(const float* __restrict__ disp,
                 const float* __restrict__ w1,
                 const float* __restrict__ w1b,
                 __bf16* __restrict__ act) {
    __shared__ __align__(16) unsigned char smem[BM * LDC * 4]; // 67584 B
    __bf16* sa0 = (__bf16*)smem;
    __bf16* sb0 = (__bf16*)(smem + TILEB / 2);        // stage0 B: +5120 B
    __bf16* sa1 = (__bf16*)(smem + TILEB);            // stage1 A
    __bf16* sb1 = (__bf16*)(smem + TILEB + TILEB / 2);
    float*  sc  = (float*)smem;

    const int tid  = threadIdx.x;
    const int lane = tid & 31;
    const int wid  = tid >> 5;      // 8 waves
    const int wr   = wid >> 1;      // 0..3 : 32-row strip
    const int wc   = wid & 1;       // 0..1 : 64-col strip
    const int l15  = lane & 15;
    const int kb8  = (lane >> 4) << 3;

    const int f0 = blockIdx.x * BN;       // gate_up column tile
    const int r0 = blockIdx.y * BM;       // token row tile
    const int e  = blockIdx.z;
    const int c  = r0 / 512;              // dispatch chunk (tile never straddles)
    const int m0 = r0 % 512;

    const float* Abase = disp + (((size_t)c * E_ + e) * 512 + m0) * H_;
    const float* Bbase = w1 + (size_t)e * H_ * F2_ + f0;

    v8f acc[2][4] = {};

    k1_load_tiles(Abase, Bbase, 0, sa0, sb0, tid);    // prologue: tile 0 -> stage 0

    const int NK = H_ / BK;   // 64
    for (int ki = 0; ki < NK; ++ki) {
        __syncthreads();
        __bf16* ca = (ki & 1) ? sa1 : sa0;
        __bf16* cb = (ki & 1) ? sb1 : sb0;
        if (ki + 1 < NK) {
            __bf16* na = (ki & 1) ? sa0 : sa1;
            __bf16* nb = (ki & 1) ? sb0 : sb1;
            k1_load_tiles(Abase, Bbase, (ki + 1) * BK, na, nb, tid);
        }
        v16bf af[2], bfr[4];
        #pragma unroll
        for (int mi = 0; mi < 2; ++mi)
            af[mi] = frag_ld(ca + (wr * 32 + mi * 16 + l15) * LDT + kb8);
        #pragma unroll
        for (int ni = 0; ni < 4; ++ni)
            bfr[ni] = frag_ld(cb + (wc * 64 + ni * 16 + l15) * LDT + kb8);
        #pragma unroll
        for (int mi = 0; mi < 2; ++mi)
            #pragma unroll
            for (int ni = 0; ni < 4; ++ni)
                acc[mi][ni] = __builtin_amdgcn_wmma_f32_16x16x32_bf16(
                    false, af[mi], false, bfr[ni], (short)0, acc[mi][ni],
                    false, false);
    }

    // Stage C fragments to LDS (C layout: VGPR v -> M = v + 8*(lane>>4), N = lane&15)
    __syncthreads();
    #pragma unroll
    for (int mi = 0; mi < 2; ++mi)
        #pragma unroll
        for (int ni = 0; ni < 4; ++ni) {
            int rb  = wr * 32 + mi * 16 + ((lane >> 4) << 3);
            int col = wc * 64 + ni * 16 + l15;
            #pragma unroll
            for (int v = 0; v < 8; ++v)
                sc[(rb + v) * LDC + col] = acc[mi][ni][v];
        }
    __syncthreads();

    // SwiGLU epilogue: 128 rows x 64 activated cols (pairs of even/odd f)
    const int i0 = f0 >> 1;
    __bf16* obase = act + ((size_t)e * MTOT + r0) * I_ + i0;
    const float* bg = w1b + (size_t)e * F2_ + f0;
    #pragma unroll
    for (int it = 0; it < 32; ++it) {
        int idx = it * 256 + tid;
        int row = idx >> 6, icol = idx & 63;
        float g = sc[row * LDC + 2 * icol]     + bg[2 * icol];
        float u = sc[row * LDC + 2 * icol + 1] + bg[2 * icol + 1];
        float a = (g / (1.f + __expf(-g))) * u;   // silu(g) * u
        obase[(size_t)row * I_ + icol] = (__bf16)a;
    }
}

// ---------------- GEMM2 + bias ----------------

__device__ __forceinline__ void k2_load_tiles(const __bf16* __restrict__ Abase,
                                              const float* __restrict__ Bbase,
                                              int kb, __bf16* sa, __bf16* sb,
                                              int tid) {
    // A tile: 128x32 bf16, pure copy -> async DMA to LDS (ASYNCcnt path).
    #pragma unroll
    for (int i = 0; i < 2; ++i) {
        int seg = i * 256 + tid;
        int row = seg >> 2, c8 = (seg & 3) << 3;
        const __bf16* g = Abase + (size_t)row * I_ + kb + c8;
        unsigned lds_off = (unsigned)(unsigned long long)(sa + row * LDT + c8);
        asm volatile("global_load_async_to_lds_b128 %0, %1, off"
                     :: "v"(lds_off), "v"(g) : "memory");
    }
    // B tile: 32x128 fp32, transpose-convert into sb[n][k] (VALU cvt path)
    #pragma unroll
    for (int i = 0; i < 4; ++i) {
        int seg = i * 256 + tid;
        int kr = seg >> 5, hi = (seg & 31) << 2;
        const float4 f = *(const float4*)(Bbase + (size_t)(kb + kr) * H_ + hi);
        sb[(hi + 0) * LDT + kr] = (__bf16)f.x;
        sb[(hi + 1) * LDT + kr] = (__bf16)f.y;
        sb[(hi + 2) * LDT + kr] = (__bf16)f.z;
        sb[(hi + 3) * LDT + kr] = (__bf16)f.w;
    }
}

__global__ __launch_bounds__(256) void
moe_gemm2(const __bf16* __restrict__ act,
          const float* __restrict__ w2,
          const float* __restrict__ w2b,
          float* __restrict__ out) {
    __shared__ __align__(16) unsigned char smem[BM * LDC * 4];
    __bf16* sa0 = (__bf16*)smem;
    __bf16* sb0 = (__bf16*)(smem + TILEB / 2);
    __bf16* sa1 = (__bf16*)(smem + TILEB);
    __bf16* sb1 = (__bf16*)(smem + TILEB + TILEB / 2);
    float*  sc  = (float*)smem;

    const int tid  = threadIdx.x;
    const int lane = tid & 31;
    const int wid  = tid >> 5;
    const int wr   = wid >> 1;
    const int wc   = wid & 1;
    const int l15  = lane & 15;
    const int kb8  = (lane >> 4) << 3;

    const int h0 = blockIdx.x * BN;
    const int r0 = blockIdx.y * BM;
    const int e  = blockIdx.z;

    const __bf16* Abase = act + ((size_t)e * MTOT + r0) * I_;
    const float*  Bbase = w2 + (size_t)e * I_ * H_ + h0;

    v8f acc[2][4] = {};

    k2_load_tiles(Abase, Bbase, 0, sa0, sb0, tid);

    const int NK = I_ / BK;   // 64
    for (int ki = 0; ki < NK; ++ki) {
        // this wave's async copies into the stage about to be consumed must
        // have landed in LDS before the block-wide barrier
        asm volatile("s_wait_asynccnt 0x0" ::: "memory");
        __syncthreads();
        __bf16* ca = (ki & 1) ? sa1 : sa0;
        __bf16* cb = (ki & 1) ? sb1 : sb0;
        if (ki + 1 < NK) {
            __bf16* na = (ki & 1) ? sa0 : sa1;
            __bf16* nb = (ki & 1) ? sb0 : sb1;
            k2_load_tiles(Abase, Bbase, (ki + 1) * BK, na, nb, tid);
        }
        v16bf af[2], bfr[4];
        #pragma unroll
        for (int mi = 0; mi < 2; ++mi)
            af[mi] = frag_ld(ca + (wr * 32 + mi * 16 + l15) * LDT + kb8);
        #pragma unroll
        for (int ni = 0; ni < 4; ++ni)
            bfr[ni] = frag_ld(cb + (wc * 64 + ni * 16 + l15) * LDT + kb8);
        #pragma unroll
        for (int mi = 0; mi < 2; ++mi)
            #pragma unroll
            for (int ni = 0; ni < 4; ++ni)
                acc[mi][ni] = __builtin_amdgcn_wmma_f32_16x16x32_bf16(
                    false, af[mi], false, bfr[ni], (short)0, acc[mi][ni],
                    false, false);
    }

    __syncthreads();
    #pragma unroll
    for (int mi = 0; mi < 2; ++mi)
        #pragma unroll
        for (int ni = 0; ni < 4; ++ni) {
            int rb  = wr * 32 + mi * 16 + ((lane >> 4) << 3);
            int col = wc * 64 + ni * 16 + l15;
            #pragma unroll
            for (int v = 0; v < 8; ++v)
                sc[(rb + v) * LDC + col] = acc[mi][ni][v];
        }
    __syncthreads();

    // Bias + coalesced fp32 store; out row r = c*512+m matches token order.
    const float* bb = w2b + (size_t)e * H_ + h0;
    float* ob = out + ((size_t)e * MTOT + r0) * H_ + h0;
    #pragma unroll
    for (int it = 0; it < 16; ++it) {
        int seg = it * 256 + tid;
        int row = seg >> 5, c4 = (seg & 31) << 2;
        float4 r;
        r.x = sc[row * LDC + c4 + 0] + bb[c4 + 0];
        r.y = sc[row * LDC + c4 + 1] + bb[c4 + 1];
        r.z = sc[row * LDC + c4 + 2] + bb[c4 + 2];
        r.w = sc[row * LDC + c4 + 3] + bb[c4 + 3];
        *(float4*)(ob + (size_t)row * H_ + c4) = r;
    }
}

extern "C" void kernel_launch(void* const* d_in, const int* in_sizes, int n_in,
                              void* d_out, int out_size, void* d_ws, size_t ws_size,
                              hipStream_t stream) {
    (void)in_sizes; (void)n_in; (void)out_size; (void)ws_size;
    const float* disp = (const float*)d_in[0];   // (1,8,8,512,2048) f32
    const float* w1   = (const float*)d_in[1];   // (8,2048,4096) f32
    const float* w1b  = (const float*)d_in[2];   // (8,4096) f32
    const float* w2   = (const float*)d_in[3];   // (8,2048,2048) f32
    const float* w2b  = (const float*)d_in[4];   // (8,2048) f32
    // d_in[5] sparsity_remap (unused by reference math), d_in[6]=BD, d_in[7]=S

    __bf16* actws = (__bf16*)d_ws;   // E*4096*2048 bf16 = 128 MiB scratch

    moe_gemm1_swiglu<<<dim3(F2_ / BN, MTOT / BM, E_), 256, 0, stream>>>(
        disp, w1, w1b, actws);
    moe_gemm2<<<dim3(H_ / BN, MTOT / BM, E_), 256, 0, stream>>>(
        actws, w2, w2b, (float*)d_out);
}
